// CongressGAT_64441689309850
// MI455X (gfx1250) — matrix operand order repaired and checked by reading, not compiled, not run on hardware
//
#include <hip/hip_runtime.h>
#include <hip/hip_bf16.h>
#include <math.h>

typedef __attribute__((ext_vector_type(16))) _Float16 v16h;
typedef __attribute__((ext_vector_type(8)))  _Float16 v8h;
typedef __attribute__((ext_vector_type(8)))  float    v8f;

#define BN_EPS 1e-5f

// ---------------- helpers ----------------
__device__ __forceinline__ unsigned fenc(float f) {
    unsigned u = __float_as_uint(f);
    return (u & 0x80000000u) ? ~u : (u | 0x80000000u);
}
__device__ __forceinline__ float fdec(unsigned u) {
    return (u & 0x80000000u) ? __uint_as_float(u & 0x7FFFFFFFu)
                             : __uint_as_float(~u);
}

// ---------------- conversion kernels ----------------
__global__ void k_f32_to_f16(const float* __restrict__ in, _Float16* __restrict__ out, size_t n) {
    size_t i = (size_t)blockIdx.x * blockDim.x + threadIdx.x;
    if (i < n) out[i] = (_Float16)in[i];
}

// W row-major [K,Nc] f32 -> Wt [Nc,K] f16 (column-major of W)
__global__ void k_transpose_f16(const float* __restrict__ W, _Float16* __restrict__ Wt, int K, int Nc) {
    int i = blockIdx.x * blockDim.x + threadIdx.x;
    if (i < K * Nc) {
        int k = i / Nc, n = i % Nc;
        Wt[(size_t)n * K + k] = (_Float16)W[i];
    }
}

__global__ void k_fill_u32(unsigned* __restrict__ p, unsigned v, size_t n) {
    size_t i = (size_t)blockIdx.x * blockDim.x + threadIdx.x;
    if (i < n) p[i] = v;
}

// block-reduce sum of x[0..n) into *out (out must be zeroed first)
__global__ void k_sum(const float* __restrict__ x, float* __restrict__ out, int n) {
    __shared__ float sm[256];
    int tid = threadIdx.x;
    float s = 0.f;
    for (size_t i = (size_t)blockIdx.x * blockDim.x + tid; i < (size_t)n;
         i += (size_t)gridDim.x * blockDim.x) s += x[i];
    sm[tid] = s;
    __syncthreads();
    for (int off = 128; off > 0; off >>= 1) {
        if (tid < off) sm[tid] += sm[tid + off];
        __syncthreads();
    }
    if (tid == 0) atomicAdd(out, sm[0]);
}

// wedot[h] = dot(We[h*C .. ], ae[h*C .. ])
__global__ void k_wedot(const float* __restrict__ We, const float* __restrict__ ae,
                        float* __restrict__ out, int H, int C) {
    int h = threadIdx.x;
    if (h < H) {
        float s = 0.f;
        for (int c = 0; c < C; ++c) s += We[h * C + c] * ae[h * C + c];
        out[h] = s;
    }
}

// ---------------- WMMA GEMM: D[M,Nc] = A[M,K] (f16 rowmajor) * B (given as Bt[Nc,K] f16) ---------
// one wave per 16x16 output tile; M % 16 == 0, Nc % 16 == 0, K % 32 == 0.
__global__ void k_gemm_wmma(const _Float16* __restrict__ A, const _Float16* __restrict__ Bt,
                            float* __restrict__ D, int M, int K, int Nc) {
    int wave   = blockIdx.x * (blockDim.x >> 5) + (threadIdx.x >> 5);
    int ntiles = Nc >> 4;
    int total  = (M >> 4) * ntiles;
    if (wave >= total) return;                 // whole-wave uniform -> EXEC stays all-1s
    int lane = threadIdx.x & 31;
    int half = lane >> 4, l15 = lane & 15;
    int mt = wave / ntiles, nt = wave - mt * ntiles;

    // A fragment: row M = l15; K chunks [8*half, 8*half+8) and [16+8*half, ...+8)
    const _Float16* arow = A  + (size_t)(mt * 16 + l15) * K + 8 * half;
    // B fragment: col N = l15; K chunks [8*half, 8*half+8) and [16+8*half, ...+8)
    const _Float16* bcol = Bt + (size_t)(nt * 16 + l15) * K + 8 * half;

    v8f c = {};
    for (int k0 = 0; k0 < K; k0 += 32) {
        v8h a0 = *(const v8h*)(arow + k0);
        v8h a1 = *(const v8h*)(arow + k0 + 16);
        v8h b0 = *(const v8h*)(bcol + k0);
        v8h b1 = *(const v8h*)(bcol + k0 + 16);
        v16h a, b;
#pragma unroll
        for (int i = 0; i < 8; ++i) {
            a[i] = a0[i]; a[8 + i] = a1[i];
            b[i] = b0[i]; b[8 + i] = b1[i];
        }
        c = __builtin_amdgcn_wmma_f32_16x16x32_f16(false, a, false, b, (short)0, c, false, false);
    }
#pragma unroll
    for (int v = 0; v < 8; ++v) {
        int row = mt * 16 + v + 8 * half;
        D[(size_t)row * Nc + nt * 16 + l15] = c[v];
    }
}

// ---------------- GAT pieces ----------------
// als[n,h] = sum_c xs[n,h,c]*as[h,c]; ald similarly
__global__ void k_node_alpha(const float* __restrict__ xs, const float* __restrict__ as,
                             const float* __restrict__ ad, float* __restrict__ als,
                             float* __restrict__ ald, int n, int H, int C) {
    int i = blockIdx.x * blockDim.x + threadIdx.x;
    if (i >= n * H) return;
    int h = i % H;
    const float* xr = xs + (size_t)(i / H) * H * C + h * C;
    float s = 0.f, d = 0.f;
    for (int c = 0; c < C; ++c) {
        float v = xr[c];
        s += v * as[h * C + c];
        d += v * ad[h * C + c];
    }
    als[i] = s;
    ald[i] = d;
}

// pass 1: alpha = leaky_relu(als[src]+ald[dst]+ea*wedot); atomic segment max over dst
__global__ void k_edge_alpha(const int* __restrict__ ei, const float* __restrict__ eattr,
                             const float* __restrict__ als, const float* __restrict__ ald,
                             const float* __restrict__ wedot, const float* __restrict__ easum,
                             float* __restrict__ alpha, unsigned* __restrict__ mx,
                             int H, int E, int EN, float invE) {
    size_t t = (size_t)blockIdx.x * blockDim.x + threadIdx.x;
    if (t >= (size_t)EN * H) return;
    int e = (int)(t / H), h = (int)(t % H);
    int s, d; float ea;
    if (e < E) { s = ei[e]; d = ei[E + e]; ea = eattr[e]; }
    else       { s = d = e - E;            ea = easum[0] * invE; }   // mean-attr self loop
    float a = als[(size_t)s * H + h] + ald[(size_t)d * H + h] + ea * wedot[h];
    a = (a > 0.f) ? a : 0.2f * a;
    alpha[t] = a;
    atomicMax(&mx[(size_t)d * H + h], fenc(a));
}

// pass 2: ex = exp(alpha - mx[dst]); atomic segment sum into den; alpha <- ex
__global__ void k_edge_exp(const int* __restrict__ ei, float* __restrict__ alpha,
                           const unsigned* __restrict__ mx, float* __restrict__ den,
                           int H, int E, int EN) {
    size_t t = (size_t)blockIdx.x * blockDim.x + threadIdx.x;
    if (t >= (size_t)EN * H) return;
    int e = (int)(t / H), h = (int)(t % H);
    int d = (e < E) ? ei[E + e] : e - E;
    float ex = __expf(alpha[t] - fdec(mx[(size_t)d * H + h]));
    alpha[t] = ex;
    atomicAdd(&den[(size_t)d * H + h], ex);
}

// pass 3: out[dst,h,c] += xs[src,h,c] * ex/(den[dst,h]+eps)
__global__ void k_edge_agg(const int* __restrict__ ei, const float* __restrict__ ex,
                           const float* __restrict__ den, const float* __restrict__ xs,
                           float* __restrict__ out, int H, int C, int E, int EN) {
    size_t t = (size_t)blockIdx.x * blockDim.x + threadIdx.x;
    int F = H * C;
    if (t >= (size_t)EN * F) return;
    size_t e = t / F; int j = (int)(t % F); int h = j / C;
    int s, d;
    if (e < (size_t)E) { s = ei[e]; d = ei[E + e]; }
    else               { s = d = (int)e - E; }
    float a = ex[e * H + h] / (den[(size_t)d * H + h] + 1e-16f);
    atomicAdd(&out[(size_t)d * F + j], xs[(size_t)s * F + j] * a);
}

// y = elu((in + bias)*gamma/sqrt(1+eps) + beta); optional f32 and f16 outputs
__global__ void k_bn_elu(const float* __restrict__ in, const float* __restrict__ bias,
                         const float* __restrict__ gamma, const float* __restrict__ beta,
                         float* __restrict__ outf, _Float16* __restrict__ outh, int n, int F) {
    size_t i = (size_t)blockIdx.x * blockDim.x + threadIdx.x;
    if (i >= (size_t)n * F) return;
    int c = (int)(i % F);
    float y = (in[i] + bias[c]) * (gamma[c] * rsqrtf(1.0f + BN_EPS)) + beta[c];
    y = (y > 0.f) ? y : expm1f(y);
    if (outf) outf[i] = y;
    if (outh) outh[i] = (_Float16)y;
}

// classifier + regressor heads (C=32 -> 16 -> {2,1})
__global__ void k_heads(const float* __restrict__ h,
                        const float* __restrict__ Wc1, const float* __restrict__ bc1,
                        const float* __restrict__ Wc2, const float* __restrict__ bc2,
                        const float* __restrict__ Wr1, const float* __restrict__ br1,
                        const float* __restrict__ Wr2, const float* __restrict__ br2,
                        float* __restrict__ cls, float* __restrict__ reg, int n) {
    int i = blockIdx.x * blockDim.x + threadIdx.x;
    if (i >= n) return;
    const float* hr = h + (size_t)i * 32;
    float t[16];
#pragma unroll
    for (int j = 0; j < 16; ++j) {
        float s = bc1[j];
        for (int c = 0; c < 32; ++c) s += hr[c] * Wc1[c * 16 + j];
        t[j] = s > 0.f ? s : 0.f;
    }
    float c0 = bc2[0], c1 = bc2[1];
#pragma unroll
    for (int j = 0; j < 16; ++j) { c0 += t[j] * Wc2[j * 2]; c1 += t[j] * Wc2[j * 2 + 1]; }
    cls[2 * i] = c0; cls[2 * i + 1] = c1;
#pragma unroll
    for (int j = 0; j < 16; ++j) {
        float s = br1[j];
        for (int c = 0; c < 32; ++c) s += hr[c] * Wr1[c * 16 + j];
        t[j] = s > 0.f ? s : 0.f;
    }
    float r = br2[0];
#pragma unroll
    for (int j = 0; j < 16; ++j) r += t[j] * Wr2[j];
    reg[i] = r;
}

// ---------------- launcher ----------------
static inline size_t align_up(size_t x) { return (x + 255) & ~(size_t)255; }

extern "C" void kernel_launch(void* const* d_in, const int* in_sizes, int n_in,
                              void* d_out, int out_size, void* d_ws, size_t ws_size,
                              hipStream_t stream) {
    const int N = 50000, E = 1600000, IN = 256, C = 32, H = 4, F1 = H * C; // 128
    const int EN = N + E;

    const float* x   = (const float*)d_in[0];
    const int*   ei  = (const int*)  d_in[1];
    const float* ea  = (const float*)d_in[2];
    const float* W1  = (const float*)d_in[3];
    const float* as1 = (const float*)d_in[4];
    const float* ad1 = (const float*)d_in[5];
    const float* We1 = (const float*)d_in[6];
    const float* ae1 = (const float*)d_in[7];
    const float* b1  = (const float*)d_in[8];
    const float* g1  = (const float*)d_in[9];
    const float* be1 = (const float*)d_in[10];
    const float* W2  = (const float*)d_in[11];
    const float* as2 = (const float*)d_in[12];
    const float* ad2 = (const float*)d_in[13];
    const float* We2 = (const float*)d_in[14];
    const float* ae2 = (const float*)d_in[15];
    const float* b2  = (const float*)d_in[16];
    const float* g2  = (const float*)d_in[17];
    const float* be2 = (const float*)d_in[18];
    const float* Wc1 = (const float*)d_in[19];
    const float* bc1 = (const float*)d_in[20];
    const float* Wc2 = (const float*)d_in[21];
    const float* bc2 = (const float*)d_in[22];
    const float* Wr1 = (const float*)d_in[23];
    const float* br1 = (const float*)d_in[24];
    const float* Wr2 = (const float*)d_in[25];
    const float* br2 = (const float*)d_in[26];

    // output layout: cls[N,2] | reg[N] | h[N,32]
    float* o_cls = (float*)d_out;
    float* o_reg = o_cls + (size_t)2 * N;
    float* o_h   = o_reg + (size_t)N;

    // workspace carve-out
    char* ws = (char*)d_ws;
    size_t off = 0;
    auto take = [&](size_t bytes) { char* p = ws + off; off += align_up(bytes); return p; };
    _Float16* xh    = (_Float16*)take((size_t)N * IN * 2);
    _Float16* w1t   = (_Float16*)take((size_t)F1 * IN * 2);
    _Float16* w2t   = (_Float16*)take((size_t)C * F1 * 2);
    _Float16* h1h   = (_Float16*)take((size_t)N * F1 * 2);
    float*    xs1   = (float*)   take((size_t)N * F1 * 4);   // reused as xs2
    float*    out1  = (float*)   take((size_t)N * F1 * 4);   // reused as out2
    float*    alpha = (float*)   take((size_t)EN * H * 4);   // reused for layer 2
    float*    als   = (float*)   take((size_t)N * H * 4);
    float*    ald   = (float*)   take((size_t)N * H * 4);
    unsigned* mx    = (unsigned*)take((size_t)N * H * 4);
    float*    den   = (float*)   take((size_t)N * H * 4);
    float*    wedot = (float*)   take(4 * 4);
    float*    easum = (float*)   take(4);
    (void)ws_size; (void)n_in; (void)in_sizes; (void)out_size;

    const int B = 256;
    const unsigned ENC_NEG_INF = 0x007FFFFFu; // fenc(-inf)
    auto blocks = [](size_t n, int b) { return (unsigned)((n + b - 1) / b); };

    // ---- prep ----
    hipMemsetAsync(easum, 0, 4, stream);
    k_sum<<<512, B, 0, stream>>>(ea, easum, E);
    k_f32_to_f16<<<blocks((size_t)N * IN, B), B, 0, stream>>>(x, xh, (size_t)N * IN);
    k_transpose_f16<<<blocks((size_t)IN * F1, B), B, 0, stream>>>(W1, w1t, IN, F1);
    k_transpose_f16<<<blocks((size_t)F1 * C, B), B, 0, stream>>>(W2, w2t, F1, C);
    k_wedot<<<1, 32, 0, stream>>>(We1, ae1, wedot, H, C);
    k_wedot<<<1, 32, 0, stream>>>(We2, ae2, wedot + H, 1, C); // layer-2 dot at wedot[H]

    // ---- layer 1: GAT(IN -> 4x32, concat) ----
    {
        int waves = (N / 16) * (F1 / 16);               // 25000
        k_gemm_wmma<<<blocks((size_t)waves * 32, B), B, 0, stream>>>(xh, w1t, xs1, N, IN, F1);
        k_node_alpha<<<blocks((size_t)N * H, B), B, 0, stream>>>(xs1, as1, ad1, als, ald, N, H, C);
        k_fill_u32<<<blocks((size_t)N * H, B), B, 0, stream>>>(mx, ENC_NEG_INF, (size_t)N * H);
        hipMemsetAsync(den, 0, (size_t)N * H * 4, stream);
        k_edge_alpha<<<blocks((size_t)EN * H, B), B, 0, stream>>>(ei, ea, als, ald, wedot, easum,
                                                                  alpha, mx, H, E, EN, 1.0f / E);
        k_edge_exp<<<blocks((size_t)EN * H, B), B, 0, stream>>>(ei, alpha, mx, den, H, E, EN);
        hipMemsetAsync(out1, 0, (size_t)N * F1 * 4, stream);
        k_edge_agg<<<blocks((size_t)EN * F1, B), B, 0, stream>>>(ei, alpha, den, xs1, out1, H, C, E, EN);
        k_bn_elu<<<blocks((size_t)N * F1, B), B, 0, stream>>>(out1, b1, g1, be1,
                                                              (float*)nullptr, h1h, N, F1);
    }

    // ---- layer 2: GAT(128 -> 32, heads=1, concat=False) ----
    {
        int waves = (N / 16) * (C / 16);                // 6250
        k_gemm_wmma<<<blocks((size_t)waves * 32, B), B, 0, stream>>>(h1h, w2t, xs1, N, F1, C);
        k_node_alpha<<<blocks((size_t)N, B), B, 0, stream>>>(xs1, as2, ad2, als, ald, N, 1, C);
        k_fill_u32<<<blocks((size_t)N, B), B, 0, stream>>>(mx, ENC_NEG_INF, (size_t)N);
        hipMemsetAsync(den, 0, (size_t)N * 4, stream);
        k_edge_alpha<<<blocks((size_t)EN, B), B, 0, stream>>>(ei, ea, als, ald, wedot + H, easum,
                                                              alpha, mx, 1, E, EN, 1.0f / E);
        k_edge_exp<<<blocks((size_t)EN, B), B, 0, stream>>>(ei, alpha, mx, den, 1, E, EN);
        hipMemsetAsync(out1, 0, (size_t)N * C * 4, stream);
        k_edge_agg<<<blocks((size_t)EN * C, B), B, 0, stream>>>(ei, alpha, den, xs1, out1, 1, C, E, EN);
        k_bn_elu<<<blocks((size_t)N * C, B), B, 0, stream>>>(out1, b2, g2, be2,
                                                             o_h, (_Float16*)nullptr, N, C);
    }

    // ---- heads ----
    k_heads<<<blocks((size_t)N, B), B, 0, stream>>>(o_h, Wc1, bc1, Wc2, bc2,
                                                    Wr1, br1, Wr2, br2, o_cls, o_reg, N);
}
